// SiamesePatchcoreModel_1314259992917
// MI455X (gfx1250) — matrix operand order repaired and checked by reading, not compiled, not run on hardware
//
#include <hip/hip_runtime.h>
#include <math.h>

// ---------------- problem constants (fixed by setup_inputs) ----------------
#define N_EMB   6272
#define M_BANK  32768
#define DIM     128
#define BATCH   8
#define PPB     784          // patches per batch image (28*28)
#define WPIX    28
#define OUTPIX  224
#define KNN     9
#define KSIZE   33
#define KHALF   16
#define RPB     32           // embedding rows per block (2 row-tiles per wave)
#define CHUNK   128          // memory-bank rows staged per iteration (32 KB bf16)
#define NITER   (M_BANK / CHUNK)

typedef __bf16 bf16x16 __attribute__((ext_vector_type(16)));
typedef float  f32x8   __attribute__((ext_vector_type(8)));

#if defined(__HIP_DEVICE_COMPILE__) && __has_builtin(__builtin_amdgcn_global_load_async_to_lds_b128)
#define USE_ASYNC 1
#else
#define USE_ASYNC 0
#endif

__device__ __forceinline__ void wait_async0() {
#if USE_ASYNC
#if __has_builtin(__builtin_amdgcn_s_wait_asynccnt)
    __builtin_amdgcn_s_wait_asynccnt(0);
#else
    asm volatile("s_wait_asynccnt 0x0" ::: "memory");
#endif
#endif
}

#if USE_ASYNC
// Builtin signature (probe-confirmed via diagnostic): void(v4i AS1*, v4i AS3*, imm, imm)
typedef int v4i __attribute__((ext_vector_type(4)));
typedef __attribute__((address_space(1))) v4i g_v4i;
typedef __attribute__((address_space(3))) v4i l_v4i;

// Stage one 32 KB chunk (128 rows x 128 dims bf16) global -> LDS via async DMA.
// 256 threads x 8 x 16B; tracked with ASYNCcnt, no VGPR round trip.
__device__ __forceinline__ void stage_chunk_async(const unsigned short* g,
                                                  unsigned short* lds, int tid) {
    g_v4i* gp = (g_v4i*)g + tid;
    l_v4i* lp = (l_v4i*)lds + tid;
    #pragma unroll
    for (int j = 0; j < 8; ++j) {
        __builtin_amdgcn_global_load_async_to_lds_b128(gp + j * 256, lp + j * 256, 0, 0);
    }
}
#endif

// ---------------- helpers ----------------
__device__ __forceinline__ unsigned short f32_to_bf16(float f) {
    unsigned int u = __float_as_uint(f);
    u += 0x7FFFu + ((u >> 16) & 1u);      // round-to-nearest-even
    return (unsigned short)(u >> 16);
}

// ============================================================================
// Kernel 1: convert rows to bf16 + compute squared norms. One wave per row.
// ============================================================================
__global__ void __launch_bounds__(256)
prep_bf16_norm_kernel(const float* __restrict__ emb, const float* __restrict__ mb,
                      unsigned short* __restrict__ embBf, unsigned short* __restrict__ mbBf,
                      float* __restrict__ x2, float* __restrict__ y2)
{
    const int wave = threadIdx.x >> 5;
    const int lane = threadIdx.x & 31;
    const int row  = blockIdx.x * 8 + wave;   // 39040 rows total

    const float* src;
    unsigned short* dst;
    float* nrm;
    if (row < N_EMB) {
        src = emb + (size_t)row * DIM; dst = embBf + (size_t)row * DIM; nrm = x2 + row;
    } else {
        int r = row - N_EMB;
        if (r >= M_BANK) return;
        src = mb + (size_t)r * DIM; dst = mbBf + (size_t)r * DIM; nrm = y2 + r;
    }

    float4 v = reinterpret_cast<const float4*>(src)[lane];   // 4 elems per lane
    float s = v.x * v.x + v.y * v.y + v.z * v.z + v.w * v.w;

    ushort4 o;
    o.x = f32_to_bf16(v.x); o.y = f32_to_bf16(v.y);
    o.z = f32_to_bf16(v.z); o.w = f32_to_bf16(v.w);
    reinterpret_cast<ushort4*>(dst)[lane] = o;

    #pragma unroll
    for (int off = 16; off >= 1; off >>= 1) s += __shfl_xor(s, off, 32);
    if (lane == 0) *nrm = s;
}

// ============================================================================
// Kernel 2: fused WMMA distance GEMM + per-row min/argmin.
// Block = 256 threads (8 waves), 32-row embedding strip (2 row-tiles / wave).
// Memory bank streamed in 128-row LDS chunks, double-buffered with async DMA.
// ============================================================================
__global__ void __launch_bounds__(256)
patch_min_wmma_kernel(const unsigned short* __restrict__ embBf,
                      const unsigned short* __restrict__ mbBf,
                      const float* __restrict__ x2, const float* __restrict__ y2,
                      float* __restrict__ ps, int* __restrict__ loc)
{
    __shared__ alignas(32) unsigned short aLds[RPB * DIM];            // 8 KB
#if USE_ASYNC
    __shared__ alignas(32) unsigned short bLds[2][CHUNK * DIM];       // 64 KB
#else
    __shared__ alignas(32) unsigned short bLds[1][CHUNK * DIM];       // 32 KB
#endif
    __shared__ float rMinV[8][RPB];
    __shared__ int   rMinI[8][RPB];

    const int tid  = threadIdx.x;
    const int wave = tid >> 5;
    const int lane = tid & 31;
    const int h    = lane >> 4;      // lane half: 0 or 1
    const int ln   = lane & 15;      // position within half (N column / M row idx)
    const int r0   = blockIdx.x * RPB;

    // ---- stage A strip (32 rows x 128 dims, bf16 = 8 KB) into LDS ----
    {
        const uint4* g = reinterpret_cast<const uint4*>(embBf + (size_t)r0 * DIM);
        uint4* s = reinterpret_cast<uint4*>(aLds);
        s[tid]       = g[tid];
        s[tid + 256] = g[tid + 256];
    }
    __syncthreads();

#if USE_ASYNC
    stage_chunk_async(mbBf, bLds[0], tid);    // prologue: chunk 0 in flight
#endif

    // ---- build A fragments in registers (contiguous K order, same as B):
    //      VGPR0..3 : K = h*16 + {0..7};  VGPR4..7 : K = h*16 + 8 + {0..7}
    bf16x16 afrag0[4], afrag1[4];
    {
        const __bf16* row0 = reinterpret_cast<const __bf16*>(aLds) + ln * DIM;
        const __bf16* row1 = reinterpret_cast<const __bf16*>(aLds) + (16 + ln) * DIM;
        #pragma unroll
        for (int c = 0; c < 4; ++c) {
            const int kb = c * 32;
            #pragma unroll
            for (int i = 0; i < 4; ++i) {
                int k0 = kb + h * 16 + 2 * i;
                afrag0[c][2 * i]     = row0[k0];
                afrag0[c][2 * i + 1] = row0[k0 + 1];
                afrag1[c][2 * i]     = row1[k0];
                afrag1[c][2 * i + 1] = row1[k0 + 1];
            }
            #pragma unroll
            for (int i = 0; i < 4; ++i) {
                int k0 = kb + h * 16 + 8 + 2 * i;
                afrag0[c][8 + 2 * i]     = row0[k0];
                afrag0[c][8 + 2 * i + 1] = row0[k0 + 1];
                afrag1[c][8 + 2 * i]     = row1[k0];
                afrag1[c][8 + 2 * i + 1] = row1[k0 + 1];
            }
        }
    }

    // per-lane row norms: C-layout row M = v + 8*h
    float xr0[8], xr1[8];
    #pragma unroll
    for (int v = 0; v < 8; ++v) {
        xr0[v] = x2[r0 + 8 * h + v];
        xr1[v] = x2[r0 + 16 + 8 * h + v];
    }

    float mv0[8], mv1[8];
    int   mi0[8], mi1[8];
    #pragma unroll
    for (int v = 0; v < 8; ++v) {
        mv0[v] = 3.4e38f; mi0[v] = 0;
        mv1[v] = 3.4e38f; mi1[v] = 0;
    }

    // software-pipelined y2 scalar (one iteration ahead; latency hides under WMMAs)
    float yv_next = y2[wave * 16 + ln];

    // ---- stream memory bank ----
    for (int it = 0; it < NITER; ++it) {
#if USE_ASYNC
        const int cur = it & 1;
        wait_async0();          // my DMA for chunk `it` done
        __syncthreads();        // everyone's DMA done; prev compute done
        if (it + 1 < NITER)
            stage_chunk_async(mbBf + (size_t)(it + 1) * CHUNK * DIM, bLds[cur ^ 1], tid);
        const unsigned short* bbuf = bLds[cur];
#else
        __syncthreads();        // previous chunk fully consumed
        {
            const uint4* g = reinterpret_cast<const uint4*>(mbBf + (size_t)it * CHUNK * DIM);
            uint4* s = reinterpret_cast<uint4*>(bLds[0]);
            #pragma unroll
            for (int j = 0; j < 8; ++j) s[tid + j * 256] = g[tid + j * 256];
        }
        __syncthreads();
        const unsigned short* bbuf = bLds[0];
#endif

        const int colg = it * CHUNK + wave * 16 + ln;
        const float yv = yv_next;
        const int nextIt = (it + 1 < NITER) ? (it + 1) : it;       // clamp, no branch
        yv_next = y2[nextIt * CHUNK + wave * 16 + ln];

        // B fragments: lane ln = column, 16 consecutive K at (h*16) per 32-K chunk
        const __bf16* bBase =
            reinterpret_cast<const __bf16*>(bbuf) + (wave * 16 + ln) * DIM + h * 16;
        bf16x16 bfr[4];
        #pragma unroll
        for (int c = 0; c < 4; ++c)
            bfr[c] = *reinterpret_cast<const bf16x16*>(bBase + c * 32);

        f32x8 c0 = {}, c1 = {};
        #pragma unroll
        for (int c = 0; c < 4; ++c) {
            c0 = __builtin_amdgcn_wmma_f32_16x16x32_bf16(
                     false, afrag0[c], false, bfr[c], (short)0, c0, false, false);
            c1 = __builtin_amdgcn_wmma_f32_16x16x32_bf16(
                     false, afrag1[c], false, bfr[c], (short)0, c1, false, false);
        }

        #pragma unroll
        for (int v = 0; v < 8; ++v) {
            float d0 = xr0[v] + yv - 2.0f * c0[v];
            float d1 = xr1[v] + yv - 2.0f * c1[v];
            if (d0 < mv0[v]) { mv0[v] = d0; mi0[v] = colg; }
            if (d1 < mv1[v]) { mv1[v] = d1; mi1[v] = colg; }
        }
    }
#if USE_ASYNC
    wait_async0();
#endif

    // ---- reduce across the 16 lanes of each half (each lane = distinct col set) ----
    #pragma unroll
    for (int v = 0; v < 8; ++v) {
        float a = mv0[v]; int ai = mi0[v];
        float b = mv1[v]; int bi = mi1[v];
        #pragma unroll
        for (int off = 8; off >= 1; off >>= 1) {
            float oa = __shfl_xor(a, off, 32);  int oai = __shfl_xor(ai, off, 32);
            float ob = __shfl_xor(b, off, 32);  int obi = __shfl_xor(bi, off, 32);
            if (oa < a || (oa == a && oai < ai)) { a = oa; ai = oai; }
            if (ob < b || (ob == b && obi < bi)) { b = ob; bi = obi; }
        }
        if (ln == 0) {
            rMinV[wave][8 * h + v]      = a;  rMinI[wave][8 * h + v]      = ai;
            rMinV[wave][16 + 8 * h + v] = b;  rMinI[wave][16 + 8 * h + v] = bi;
        }
    }
    __syncthreads();

    // ---- merge the 8 waves' partials, write sqrt(min d2) + argmin ----
    if (tid < RPB) {
        float mv = rMinV[0][tid]; int mi = rMinI[0][tid];
        #pragma unroll
        for (int w = 1; w < 8; ++w) {
            float ov = rMinV[w][tid]; int oi = rMinI[w][tid];
            if (ov < mv || (ov == mv && oi < mi)) { mv = ov; mi = oi; }
        }
        ps[r0 + tid]  = sqrtf(fmaxf(mv, 0.0f));
        loc[r0 + tid] = mi;
    }
}

// ============================================================================
// Kernel 3: per-image score. One block per batch image.
// ============================================================================
__global__ void __launch_bounds__(256)
image_score_kernel(const float* __restrict__ emb, const float* __restrict__ mb,
                   const float* __restrict__ ps, const int* __restrict__ loc,
                   float* __restrict__ out)
{
    __shared__ float sv[256];
    __shared__ int   si[256];
    __shared__ float nn[DIM];
    __shared__ float tv[256 * KNN];
    __shared__ int   tix[256 * KNN];
    __shared__ float shScore;
    __shared__ int   shP, shNN;

    const int b = blockIdx.x, tid = threadIdx.x;

    // --- phase 1: argmax patch score (first occurrence on ties) ---
    float bv = -3.4e38f; int bp = 0x7fffffff;
    for (int p = tid; p < PPB; p += 256) {
        float v = ps[b * PPB + p];
        if (v > bv) { bv = v; bp = p; }
    }
    sv[tid] = bv; si[tid] = bp;
    __syncthreads();
    for (int s = 128; s >= 1; s >>= 1) {
        if (tid < s) {
            float ov = sv[tid + s]; int oi = si[tid + s];
            if (ov > sv[tid] || (ov == sv[tid] && oi < si[tid])) { sv[tid] = ov; si[tid] = oi; }
        }
        __syncthreads();
    }
    if (tid == 0) { shScore = sv[0]; shP = si[0]; shNN = loc[b * PPB + si[0]]; }
    __syncthreads();

    if (tid < DIM) nn[tid] = mb[(size_t)shNN * DIM + tid];
    __syncthreads();

    // --- phase 2: per-thread top-9 smallest d2(nn_sample, bank) ---
    float t9v[KNN]; int t9i[KNN];
    #pragma unroll
    for (int j = 0; j < KNN; ++j) { t9v[j] = 3.4e38f; t9i[j] = 0x7fffffff; }
    for (int m = tid; m < M_BANK; m += 256) {
        const float* r = mb + (size_t)m * DIM;
        float d2 = 0.0f;
        for (int k = 0; k < DIM; ++k) { float d = nn[k] - r[k]; d2 += d * d; }
        if (d2 < t9v[KNN - 1]) {
            int j = KNN - 1;
            while (j > 0 && d2 < t9v[j - 1]) { t9v[j] = t9v[j - 1]; t9i[j] = t9i[j - 1]; --j; }
            t9v[j] = d2; t9i[j] = m;
        }
    }
    #pragma unroll
    for (int j = 0; j < KNN; ++j) { tv[tid * KNN + j] = t9v[j]; tix[tid * KNN + j] = t9i[j]; }
    __syncthreads();

    // --- phase 3: serial merge + re-ranked softmax weighting ---
    if (tid == 0) {
        int selI[KNN];
        for (int j = 0; j < KNN; ++j) {
            float mv = 3.4e38f; int mi = 0x7fffffff; int mpos = -1;
            for (int e = 0; e < 256 * KNN; ++e) {
                if (tv[e] < mv || (tv[e] == mv && tix[e] < mi)) { mv = tv[e]; mi = tix[e]; mpos = e; }
            }
            selI[j] = mi;
            tv[mpos] = 3.4e38f; tix[mpos] = 0x7fffffff;
        }
        const float* mf = emb + (size_t)(b * PPB + shP) * DIM;
        float dist[KNN], dmax = -3.4e38f;
        for (int j = 0; j < KNN; ++j) {
            const float* r = mb + (size_t)selI[j] * DIM;
            float d2 = 0.0f;
            for (int k = 0; k < DIM; ++k) { float d = mf[k] - r[k]; d2 += d * d; }
            dist[j] = sqrtf(fmaxf(d2, 0.0f));
            dmax = fmaxf(dmax, dist[j]);
        }
        float sum = 0.0f;
        for (int j = 0; j < KNN; ++j) sum += expf(dist[j] - dmax);
        float s0 = expf(dist[0] - dmax) / sum;
        out[b] = (1.0f - s0) * shScore;
    }
}

// ============================================================================
// Kernel 4: bilinear 28 -> 224 (half-pixel centers, edge clamp)
// ============================================================================
__global__ void __launch_bounds__(256)
resize_kernel(const float* __restrict__ ps, float* __restrict__ outr)
{
    int idx = blockIdx.x * 256 + threadIdx.x;
    if (idx >= BATCH * OUTPIX * OUTPIX) return;
    int x = idx % OUTPIX;
    int y = (idx / OUTPIX) % OUTPIX;
    int b = idx / (OUTPIX * OUTPIX);

    float sy = (y + 0.5f) * ((float)WPIX / OUTPIX) - 0.5f;
    float sx = (x + 0.5f) * ((float)WPIX / OUTPIX) - 0.5f;
    int y0 = (int)floorf(sy), x0 = (int)floorf(sx);
    float fy = sy - (float)y0, fx = sx - (float)x0;
    int y0c = min(max(y0, 0), WPIX - 1), y1c = min(max(y0 + 1, 0), WPIX - 1);
    int x0c = min(max(x0, 0), WPIX - 1), x1c = min(max(x0 + 1, 0), WPIX - 1);

    const float* p = ps + b * PPB;
    float v00 = p[y0c * WPIX + x0c], v01 = p[y0c * WPIX + x1c];
    float v10 = p[y1c * WPIX + x0c], v11 = p[y1c * WPIX + x1c];
    float v0 = v00 + (v01 - v00) * fx;
    float v1 = v10 + (v11 - v10) * fx;
    outr[idx] = v0 + (v1 - v0) * fy;
}

// ============================================================================
// Kernels 5/6: separable 33-tap sigma=4 Gaussian with reflect padding
// ============================================================================
__device__ __forceinline__ int reflect_idx(int i) {
    if (i < 0) i = -i;
    if (i >= OUTPIX) i = 2 * OUTPIX - 2 - i;
    return i;
}

__global__ void __launch_bounds__(256)
blur_v_kernel(const float* __restrict__ in, float* __restrict__ out)
{
    int idx = blockIdx.x * 256 + threadIdx.x;
    if (idx >= BATCH * OUTPIX * OUTPIX) return;
    int x = idx % OUTPIX;
    int y = (idx / OUTPIX) % OUTPIX;
    int b = idx / (OUTPIX * OUTPIX);

    float g[KSIZE], gs = 0.0f;
    for (int i = 0; i < KSIZE; ++i) {
        float t = (float)(i - KHALF) * 0.25f;
        g[i] = __expf(-0.5f * t * t);
        gs += g[i];
    }
    float inv = 1.0f / gs, acc = 0.0f;
    const float* base = in + (size_t)b * OUTPIX * OUTPIX;
    for (int i = 0; i < KSIZE; ++i) {
        int yy = reflect_idx(y + i - KHALF);
        acc += g[i] * base[yy * OUTPIX + x];
    }
    out[idx] = acc * inv;
}

__global__ void __launch_bounds__(256)
blur_h_kernel(const float* __restrict__ in, float* __restrict__ out)
{
    int idx = blockIdx.x * 256 + threadIdx.x;
    if (idx >= BATCH * OUTPIX * OUTPIX) return;
    int x = idx % OUTPIX;
    int y = (idx / OUTPIX) % OUTPIX;
    int b = idx / (OUTPIX * OUTPIX);

    float g[KSIZE], gs = 0.0f;
    for (int i = 0; i < KSIZE; ++i) {
        float t = (float)(i - KHALF) * 0.25f;
        g[i] = __expf(-0.5f * t * t);
        gs += g[i];
    }
    float inv = 1.0f / gs, acc = 0.0f;
    const float* base = in + (size_t)b * OUTPIX * OUTPIX + (size_t)y * OUTPIX;
    for (int i = 0; i < KSIZE; ++i) {
        int xx = reflect_idx(x + i - KHALF);
        acc += g[i] * base[xx];
    }
    out[idx] = acc * inv;
}

// ============================================================================
// launch
// ============================================================================
extern "C" void kernel_launch(void* const* d_in, const int* in_sizes, int n_in,
                              void* d_out, int out_size, void* d_ws, size_t ws_size,
                              hipStream_t stream)
{
    const float* emb = (const float*)d_in[0];
    const float* mb  = (const float*)d_in[1];
    float* out = (float*)d_out;

    char* ws = (char*)d_ws;
    size_t off = 0;
    auto take = [&](size_t bytes) -> char* {
        off = (off + 255) & ~(size_t)255;
        char* p = ws + off;
        off += bytes;
        return p;
    };
    unsigned short* embBf = (unsigned short*)take((size_t)N_EMB * DIM * 2);
    unsigned short* mbBf  = (unsigned short*)take((size_t)M_BANK * DIM * 2);
    float* x2   = (float*)take((size_t)N_EMB * 4);
    float* y2   = (float*)take((size_t)M_BANK * 4);
    float* ps   = (float*)take((size_t)N_EMB * 4);
    int*   loc  = (int*)  take((size_t)N_EMB * 4);
    float* rsz  = (float*)take((size_t)BATCH * OUTPIX * OUTPIX * 4);
    float* tmp  = (float*)take((size_t)BATCH * OUTPIX * OUTPIX * 4);
    (void)ws_size; (void)in_sizes; (void)n_in; (void)out_size;

    // 1) bf16 conversion + norms
    prep_bf16_norm_kernel<<<(N_EMB + M_BANK) / 8, 256, 0, stream>>>(
        emb, mb, embBf, mbBf, x2, y2);

    // 2) fused WMMA distance + min/argmin  (196 strips of 32 rows)
    patch_min_wmma_kernel<<<N_EMB / RPB, 256, 0, stream>>>(
        embBf, mbBf, x2, y2, ps, loc);

    // 3) per-image score -> out[0..7]
    image_score_kernel<<<BATCH, 256, 0, stream>>>(emb, mb, ps, loc, out);

    // 4-6) anomaly map -> out[8..]
    const int npx = BATCH * OUTPIX * OUTPIX;
    const int blk = (npx + 255) / 256;
    resize_kernel<<<blk, 256, 0, stream>>>(ps, rsz);
    blur_v_kernel<<<blk, 256, 0, stream>>>(rsz, tmp);
    blur_h_kernel<<<blk, 256, 0, stream>>>(tmp, out + BATCH);
}
